// MulitModelGraphAttentionLayer_45088566673473
// MI455X (gfx1250) — matrix operand order repaired and checked by reading, not compiled, not run on hardware
//
#include <hip/hip_runtime.h>
#include <hip/hip_bf16.h>

#define ALPHA 0.2f
#define NEGV  -9000000000000000.0f
#define EPSV  1e-5f

typedef __attribute__((ext_vector_type(16))) _Float16 v16h;
typedef __attribute__((ext_vector_type(8)))  float    v8f;

union V16H { _Float16 h[16]; v16h v; };

// ---- WMMA tile loaders (row-major f16 source) ------------------------------
// A-matrix 16x32 f16 layout (ISA 7.12.2): lane L -> row L%16;
// elements 0..7 = K = (L/16)*8 + 0..7 ; elements 8..15 = K = (L/16)*8 + 16..23
__device__ __forceinline__ v16h ldA(const _Float16* p) {
  v16h a;
  reinterpret_cast<float4*>(&a)[0] = reinterpret_cast<const float4*>(p)[0];
  reinterpret_cast<float4*>(&a)[1] = reinterpret_cast<const float4*>(p + 16)[0];
  return a;
}
// B-matrix 32x16 f16 layout: lane L -> col L%16; elements 0..7 = K = (L/16)*8 + 0..7 ; elements 8..15 = K = (L/16)*8 + 16..23
// source is transposed (col-major of B == row of BT), so each 8-half run is contiguous
__device__ __forceinline__ v16h ldB(const _Float16* p) {
  v16h b;
  reinterpret_cast<float4*>(&b)[0] = reinterpret_cast<const float4*>(p)[0];
  reinterpret_cast<float4*>(&b)[1] = reinterpret_cast<const float4*>(p + 16)[0];
  return b;
}

// ---- Kernel 1: w1a = W1 @ a1, w2a = W2 @ a2 --------------------------------
__global__ void k_wa(const float* __restrict__ W1, const float* __restrict__ W2,
                     const float* __restrict__ a, float* __restrict__ w1a,
                     float* __restrict__ w2a) {
  int i = blockIdx.x * blockDim.x + threadIdx.x;
  if (i >= 512) return;
  const float* a1 = a;
  const float* a2 = a + 512;
  float s1 = 0.f, s2 = 0.f;
  for (int k = 0; k < 512; ++k) {
    s1 += W1[i * 512 + k] * a1[k];
    s2 += W2[i * 512 + k] * a2[k];
  }
  w1a[i] = s1;
  w2a[i] = s2;
}

// ---- Kernel 2: W3T[n][k] = f16(W3[k][n]) -----------------------------------
__global__ void k_w3t(const float* __restrict__ W3, _Float16* __restrict__ W3T) {
  int idx = blockIdx.x * 256 + threadIdx.x;       // 512*512
  int k = idx >> 9, n = idx & 511;
  W3T[(size_t)n * 512 + k] = (_Float16)W3[(size_t)k * 512 + n];
}

// ---- Kernel 3: e1 = h1 . w1a / e2 = h2 . w2a (+ h2 -> f16) -----------------
__global__ void k_rowvec(const float* __restrict__ h1, const float* __restrict__ h2,
                         const float* __restrict__ w1a, const float* __restrict__ w2a,
                         float* __restrict__ e1, float* __restrict__ e2,
                         _Float16* __restrict__ h2f) {
  int row   = blockIdx.x;    // 0..8191 (b*1024 + n)
  int which = blockIdx.y;    // 0 -> h1/e1, 1 -> h2/e2
  const float* h = which ? h2 : h1;
  const float* w = which ? w2a : w1a;
  __shared__ float red[128];
  float s = 0.f;
  for (int f = threadIdx.x; f < 512; f += 128) {
    float v = h[(size_t)row * 512 + f];
    s += v * w[f];
    if (which) h2f[(size_t)row * 512 + f] = (_Float16)v;
  }
  red[threadIdx.x] = s;
  __syncthreads();
  for (int off = 64; off > 0; off >>= 1) {
    if (threadIdx.x < off) red[threadIdx.x] += red[threadIdx.x + off];
    __syncthreads();
  }
  if (threadIdx.x == 0) (which ? e2 : e1)[row] = red[0];
}

// ---- Kernel 4: per-row softmax stats (max, 1/sum) --------------------------
__global__ void k_smstats(const float* __restrict__ e1, const float* __restrict__ e2,
                          const int* __restrict__ adj, float* __restrict__ smax,
                          float* __restrict__ sinv) {
  int row = blockIdx.x;                 // b*1024 + i
  int b = row >> 10;
  float e1v = e1[row];
  const int*   arow = adj + (size_t)row * 1024;
  const float* e2b  = e2 + ((size_t)b << 10);
  __shared__ float red[256];
  float vals[4];
  float m = -3.0e38f;
#pragma unroll
  for (int t = 0; t < 4; ++t) {
    int j = threadIdx.x + t * 256;
    float e = e1v + e2b[j];
    e = e > 0.f ? e : ALPHA * e;
    e = arow[j] > 0 ? e : NEGV;
    vals[t] = e;
    m = fmaxf(m, e);
  }
  red[threadIdx.x] = m;
  __syncthreads();
  for (int off = 128; off > 0; off >>= 1) {
    if (threadIdx.x < off) red[threadIdx.x] = fmaxf(red[threadIdx.x], red[threadIdx.x + off]);
    __syncthreads();
  }
  m = red[0];
  __syncthreads();
  float s = 0.f;
#pragma unroll
  for (int t = 0; t < 4; ++t) s += __expf(vals[t] - m);
  red[threadIdx.x] = s;
  __syncthreads();
  for (int off = 128; off > 0; off >>= 1) {
    if (threadIdx.x < off) red[threadIdx.x] += red[threadIdx.x + off];
    __syncthreads();
  }
  if (threadIdx.x == 0) { smax[row] = m; sinv[row] = 1.0f / red[0]; }
}

// ---- Kernel 5: WvT[b][f][j] = (h2 @ W3)^T in f16 via WMMA ------------------
// One wave: one j-tile x 4 f-tiles, manually double-buffered so WMMAs consume
// tiles loaded one iteration earlier (keeps prefetch loads in flight).
__global__ void k_gemm_wv(const _Float16* __restrict__ h2f,
                          const _Float16* __restrict__ W3T,
                          _Float16* __restrict__ WvT) {
  int lane = threadIdx.x & 31;
  int wid  = (blockIdx.x * blockDim.x + threadIdx.x) >> 5;  // 4096 waves
  int fg = wid & 7;           // 8 groups of 4 f-tiles
  int jt = (wid >> 3) & 63;   // 64 j-tiles
  int b  = wid >> 9;          // 8 batches
  int r  = lane & 15;
  int hi = lane >> 4;
  const _Float16* A0 = h2f + ((size_t)b * 1024 + jt * 16 + r) * 512 + hi * 8;
  const _Float16* B0 = W3T + ((size_t)(fg * 64 + r)) * 512 + hi * 8;
  const size_t BS = (size_t)16 * 512;

  v16h av = ldA(A0);
  v16h bv[4];
#pragma unroll
  for (int q = 0; q < 4; ++q) bv[q] = ldB(B0 + q * BS);
  v8f c[4] = {v8f{}, v8f{}, v8f{}, v8f{}};
#pragma unroll
  for (int kk = 0; kk < 16; ++kk) {
    v16h an;
    v16h bn[4];
    if (kk < 15) {
      int kn = (kk + 1) * 32;
      an = ldA(A0 + kn);
#pragma unroll
      for (int q = 0; q < 4; ++q) bn[q] = ldB(B0 + q * BS + kn);
    }
#pragma unroll
    for (int q = 0; q < 4; ++q)
      c[q] = __builtin_amdgcn_wmma_f32_16x16x32_f16(false, av, false, bv[q], (short)0,
                                                    c[q], false, false);
    if (kk < 15) {
      av = an;
#pragma unroll
      for (int q = 0; q < 4; ++q) bv[q] = bn[q];
    }
  }
  // C layout: lane -> col n=r, rows m = hi*8 + g. Store transposed: contiguous in j.
#pragma unroll
  for (int q = 0; q < 4; ++q) {
    V16H o;
#pragma unroll
    for (int g = 0; g < 8; ++g) o.h[g] = (_Float16)c[q][g];
    _Float16* out =
        WvT + ((size_t)b * 512 + fg * 64 + q * 16 + r) * 1024 + jt * 16 + hi * 8;
    reinterpret_cast<float4*>(out)[0] = reinterpret_cast<float4*>(&o)[0];
  }
}

// ---- attention-tile helpers -------------------------------------------------
__device__ __forceinline__ void att_fetch(const float* __restrict__ e2b,
                                          const int* __restrict__ arow, int jb,
                                          float4 ev[4], int4 dv[4]) {
#pragma unroll
  for (int grp = 0; grp < 2; ++grp) {
    int j0 = jb + grp * 16;
    ev[grp * 2 + 0] = *reinterpret_cast<const float4*>(e2b + j0);
    ev[grp * 2 + 1] = *reinterpret_cast<const float4*>(e2b + j0 + 4);
    dv[grp * 2 + 0] = *reinterpret_cast<const int4*>(arow + j0);
    dv[grp * 2 + 1] = *reinterpret_cast<const int4*>(arow + j0 + 4);
  }
}
__device__ __forceinline__ v16h att_build(const float4 ev[4], const int4 dv[4],
                                          float e1v, float m, float inv) {
  V16H a;
#pragma unroll
  for (int q = 0; q < 4; ++q) {
    const float* ef = reinterpret_cast<const float*>(&ev[q]);
    const int*   df = reinterpret_cast<const int*>(&dv[q]);
#pragma unroll
    for (int t = 0; t < 4; ++t) {
      float e = e1v + ef[t];
      e = e > 0.f ? e : ALPHA * e;
      e = df[t] > 0 ? e : NEGV;
      a.h[q * 4 + t] = (_Float16)(__expf(e - m) * inv);  // q*4+t == grp*8 + elem
    }
  }
  return a.v;
}

// ---- Kernel 6: h' = softmax(e) @ Wv via WMMA -------------------------------
// One wave: one i-tile x 4 f-tiles. Next iteration's e2/adj vectors and B
// tiles are prefetched before the exp-heavy attention build, so VALU work and
// XDL WMMAs cover the load latency.
__global__ void k_gemm_att(const float* __restrict__ e1, const float* __restrict__ e2,
                           const int* __restrict__ adj, const float* __restrict__ smax,
                           const float* __restrict__ sinv,
                           const _Float16* __restrict__ WvT, float* __restrict__ hp) {
  int lane = threadIdx.x & 31;
  int wid  = (blockIdx.x * blockDim.x + threadIdx.x) >> 5;  // 4096 waves
  int fg = wid & 7;           // 8 groups of 4 f-tiles
  int it = (wid >> 3) & 63;   // 64 i-tiles
  int b  = wid >> 9;          // 8 batches
  int r  = lane & 15;
  int hi = lane >> 4;
  int i    = it * 16 + r;
  int rowg = (b << 10) + i;
  float e1v = e1[rowg];
  float m   = smax[rowg];
  float inv = sinv[rowg];
  const float* e2b  = e2 + ((size_t)b << 10);
  const int*   arow = adj + (size_t)rowg * 1024;
  const _Float16* B0 = WvT + ((size_t)b * 512 + fg * 64 + r) * 1024 + hi * 8;
  const size_t BS = (size_t)16 * 1024;

  float4 ev[4];
  int4   dv[4];
  att_fetch(e2b, arow, hi * 8, ev, dv);
  v16h bv[4];
#pragma unroll
  for (int q = 0; q < 4; ++q) bv[q] = ldB(B0 + q * BS);

  v8f c[4] = {v8f{}, v8f{}, v8f{}, v8f{}};
#pragma unroll
  for (int kk = 0; kk < 32; ++kk) {
    float4 evn[4];
    int4   dvn[4];
    v16h   bn[4];
    if (kk < 31) {
      int kn = (kk + 1) * 32;
      att_fetch(e2b, arow, kn + hi * 8, evn, dvn);
#pragma unroll
      for (int q = 0; q < 4; ++q) bn[q] = ldB(B0 + q * BS + kn);
    }
    v16h av = att_build(ev, dv, e1v, m, inv);
#pragma unroll
    for (int q = 0; q < 4; ++q)
      c[q] = __builtin_amdgcn_wmma_f32_16x16x32_f16(false, av, false, bv[q], (short)0,
                                                    c[q], false, false);
    if (kk < 31) {
#pragma unroll
      for (int q = 0; q < 4; ++q) { ev[q] = evn[q]; dv[q] = dvn[q]; bv[q] = bn[q]; }
    }
  }
  // C: lane col n=r, rows m = hi*8+g -> hp[b][i][f] row-major
#pragma unroll
  for (int q = 0; q < 4; ++q) {
    float* out = hp + ((size_t)(b * 1024 + it * 16 + hi * 8)) * 512 + fg * 64 +
                 q * 16 + r;
#pragma unroll
    for (int g = 0; g < 8; ++g) out[(size_t)g * 512] = c[q][g];
  }
}

// ---- Kernel 7: layernorm + exact GELU --------------------------------------
__global__ void k_lngelu(const float* __restrict__ hp, const float* __restrict__ gamma,
                         const float* __restrict__ beta, float* __restrict__ out) {
  int row = blockIdx.x;  // 8192
  const float* x = hp + (size_t)row * 512;
  __shared__ float red[256];
  float v0 = x[threadIdx.x];
  float v1 = x[threadIdx.x + 256];
  red[threadIdx.x] = v0 + v1;
  __syncthreads();
  for (int off = 128; off > 0; off >>= 1) {
    if (threadIdx.x < off) red[threadIdx.x] += red[threadIdx.x + off];
    __syncthreads();
  }
  float mu = red[0] * (1.0f / 512.0f);
  __syncthreads();
  float d0 = v0 - mu, d1 = v1 - mu;
  red[threadIdx.x] = d0 * d0 + d1 * d1;
  __syncthreads();
  for (int off = 128; off > 0; off >>= 1) {
    if (threadIdx.x < off) red[threadIdx.x] += red[threadIdx.x + off];
    __syncthreads();
  }
  float rs = rsqrtf(red[0] * (1.0f / 512.0f) + EPSV);
  {
    int f = threadIdx.x;
    float y = d0 * rs * gamma[f] + beta[f];
    out[(size_t)row * 512 + f] = 0.5f * y * (1.0f + erff(y * 0.70710678118f));
    f += 256;
    float y2 = d1 * rs * gamma[f] + beta[f];
    out[(size_t)row * 512 + f] = 0.5f * y2 * (1.0f + erff(y2 * 0.70710678118f));
  }
}

extern "C" void kernel_launch(void* const* d_in, const int* in_sizes, int n_in,
                              void* d_out, int out_size, void* d_ws, size_t ws_size,
                              hipStream_t stream) {
  const float* h1    = (const float*)d_in[0];
  const float* h2    = (const float*)d_in[1];
  const int*   adj   = (const int*)d_in[2];
  const float* W1    = (const float*)d_in[3];
  const float* W2    = (const float*)d_in[4];
  const float* W3    = (const float*)d_in[5];
  const float* a     = (const float*)d_in[6];
  const float* gamma = (const float*)d_in[7];
  const float* beta  = (const float*)d_in[8];
  float* outp = (float*)d_out;

  char* ws = (char*)d_ws;
  size_t off = 0;
  auto alloc = [&](size_t bytes) { char* p = ws + off; off += (bytes + 255) & ~(size_t)255; return p; };
  float*    w1a  = (float*)alloc(512 * 4);
  float*    w2a  = (float*)alloc(512 * 4);
  float*    e1   = (float*)alloc(8192 * 4);
  float*    e2   = (float*)alloc(8192 * 4);
  float*    smax = (float*)alloc(8192 * 4);
  float*    sinv = (float*)alloc(8192 * 4);
  _Float16* W3T  = (_Float16*)alloc((size_t)512 * 512 * 2);
  _Float16* h2f  = (_Float16*)alloc((size_t)8192 * 512 * 2);
  _Float16* WvT  = (_Float16*)alloc((size_t)8 * 512 * 1024 * 2);
  float*    hp   = (float*)alloc((size_t)8192 * 512 * 4);

  k_wa<<<2, 256, 0, stream>>>(W1, W2, a, w1a, w2a);
  k_w3t<<<1024, 256, 0, stream>>>(W3, W3T);
  k_rowvec<<<dim3(8192, 2), 128, 0, stream>>>(h1, h2, w1a, w2a, e1, e2, h2f);
  k_smstats<<<8192, 256, 0, stream>>>(e1, e2, adj, smax, sinv);
  k_gemm_wv<<<512, 256, 0, stream>>>(h2f, W3T, WvT);
  k_gemm_att<<<512, 256, 0, stream>>>(e1, e2, adj, smax, sinv, WvT, hp);
  k_lngelu<<<8192, 256, 0, stream>>>(hp, gamma, beta, outp);
}
